// MultiHeadSelfAttention_3753801417569
// MI455X (gfx1250) — matrix operand order repaired and checked by reading, not compiled, not run on hardware
//
#include <hip/hip_runtime.h>

typedef _Float16 f16;
typedef __attribute__((ext_vector_type(16))) _Float16 v16h;
typedef __attribute__((ext_vector_type(8)))  _Float16 v8h;
typedef __attribute__((ext_vector_type(8)))  float    v8f;

#define D_MODEL 1024
#define T_SEQ   2048
#define NBATCH  2
#define NHEAD   16
#define HDIM    64
#define M_ROWS  (NBATCH * T_SEQ)   // 4096
#define N_QKV   (3 * D_MODEL)      // 3072
#define LDP     40                 // padded LDS row stride (f16 elems) = 80B

#if defined(__has_builtin)
#  if __has_builtin(__builtin_amdgcn_tensor_load_to_lds)
#    define HAVE_TDM 1
#  endif
#endif
#ifndef HAVE_TDM
#  define HAVE_TDM 0
#endif

// ---------- WMMA fragment helpers (layouts per CDNA5 ISA 7.12.2) ----------

__device__ __forceinline__ v8h ld8(const f16* p) { return *(const v8h*)p; }

// A-matrix 16x32 f16 fragment. p points at (row, c0) where c0 = half*8 within
// the 32-wide contraction chunk. elems[0..7]=p[0..7], elems[8..15]=p[16..23].
__device__ __forceinline__ v16h make_a(const f16* p) {
  v8h lo = ld8(p), hi = ld8(p + 16);
  v16h r;
#pragma unroll
  for (int i = 0; i < 8; ++i) { r[i] = lo[i]; r[i + 8] = hi[i]; }
  return r;
}

// B-matrix 32x16 f16 fragment: lane holds one column; p points at (col, c0),
// c0 = half*8. elems[0..7]=p[0..7], elems[8..15]=p[16..23] (same as A).
__device__ __forceinline__ v16h make_b(const f16* p) {
  v8h lo = ld8(p), hi = ld8(p + 16);
  v16h r;
#pragma unroll
  for (int i = 0; i < 8; ++i) { r[i] = lo[i]; r[i + 8] = hi[i]; }
  return r;
}

__device__ __forceinline__ v8f wmma16(v16h a, v16h b, v8f c) {
  return __builtin_amdgcn_wmma_f32_16x16x32_f16(false, a, false, b,
                                                (short)0, c, false, false);
}

__device__ __forceinline__ v8f vzero8() {
  v8f z = {0.f, 0.f, 0.f, 0.f, 0.f, 0.f, 0.f, 0.f};
  return z;
}

// ---------- Tensor Data Mover: 128x32 f16 tile, global -> padded LDS ----------
// D# per CDNA5 ISA §8.3/8.4. Padding: tile row = 32 f16 = 64B = 16 DWORDs
// (pad_interval code 3), pad_amount code 3 = 4 DWORDs -> LDS row stride 80B,
// identical to the LDP=40 software layout used by the WMMA fragment loads.

#if HAVE_TDM
typedef __attribute__((ext_vector_type(4))) unsigned int v4u;
typedef __attribute__((ext_vector_type(8))) int          v8i;
typedef __attribute__((ext_vector_type(4))) int          v4i;

__device__ __forceinline__ void tdm_load_tile(unsigned int lds_off,
                                              const f16* gsrc) {
  unsigned long long ga = (unsigned long long)(size_t)gsrc;
  v4u g0 = {1u,                                 // count=1, user descriptor
            lds_off,                            // LDS byte address
            (unsigned int)ga,                   // global_addr[31:0]
            (unsigned int)((ga >> 32) & 0x1FFFFFFu) | 0x80000000u};  // type=2
  v8i g1 = {
      (int)((1u << 16)        // data_size = 2 bytes
            | (1u << 20)      // pad_enable
            | (3u << 22)      // pad_interval: 16 DWORDs
            | (3u << 25)),    // pad_amount: 4 DWORDs
      (int)(32u << 16),       // tensor_dim0 = 32 (lo16 in bits 63:48)
      (int)(128u << 16),      // tensor_dim0 hi = 0 | tensor_dim1 lo16 = 128
      (int)(32u << 16),       // tensor_dim1 hi = 0 | tile_dim0 = 32
      128,                    // tile_dim1 = 128 | tile_dim2 = 0
      1024,                   // tensor_dim0_stride[31:0] = 1024 elems
      0, 0};                  // stride0 hi, tensor_dim1_stride (unused, 2D)
  v4i z4 = {0, 0, 0, 0};
#if __clang_major__ >= 23
  v8i z8 = {0, 0, 0, 0, 0, 0, 0, 0};
  __builtin_amdgcn_tensor_load_to_lds(g0, g1, z4, z4, z8, 0);
#else
  __builtin_amdgcn_tensor_load_to_lds(g0, g1, z4, z4, 0);
#endif
}
#endif  // HAVE_TDM

// ---------- precision conversion kernels ----------

__global__ void cvt_f16_kernel(const float* __restrict__ in,
                               f16* __restrict__ out, int n) {
  for (int i = blockIdx.x * blockDim.x + threadIdx.x; i < n;
       i += gridDim.x * blockDim.x)
    out[i] = (f16)in[i];
}

// in: [K][N] f32 row-major, out: [N][K] f16 row-major (transposed)
__global__ void cvt_trans_f16_kernel(const float* __restrict__ in,
                                     f16* __restrict__ out, int K, int N) {
  int total = K * N;
  for (int i = blockIdx.x * blockDim.x + threadIdx.x; i < total;
       i += gridDim.x * blockDim.x) {
    int n = i / K;
    int k = i - n * K;
    out[i] = (f16)in[(size_t)k * N + n];
  }
}

// ---------- QKV projection GEMM: [4096 x 1024] x [1024 x 3072] ----------
// Block: 128x128 tile, 8 waves (2 along M x 4 along N), wave = 64x32.
// TDM double-buffered LDS staging: wave 0 issues tensor_load_to_lds for the
// next K-slab while all waves run WMMAs on the current one.
// Epilogue: +bias, scatter to Q (scaled by 1/8), K [B,H,T,64], V^T [B,H,64,T].

__global__ __launch_bounds__(256) void gemm_qkv_kernel(
    const f16* __restrict__ X, const f16* __restrict__ WT,
    const float* __restrict__ bias, f16* __restrict__ Q, f16* __restrict__ Kt,
    f16* __restrict__ Vt) {
  __shared__ __attribute__((aligned(16))) f16 As[2][128 * LDP];
  __shared__ __attribute__((aligned(16))) f16 Bs[2][128 * LDP];
  const int m0 = blockIdx.y * 128;
  const int n0 = blockIdx.x * 128;
  const int t = threadIdx.x;
  const int lane = t & 31, wid = t >> 5;
  const int wm = wid & 1, wn = wid >> 1;
  const int half = lane >> 4, ln = lane & 15;

  v8f acc[4][2];
#pragma unroll
  for (int i = 0; i < 4; ++i)
#pragma unroll
    for (int j = 0; j < 2; ++j) acc[i][j] = vzero8();

  const f16* Arow = X + (size_t)m0 * D_MODEL;
  const f16* Brow = WT + (size_t)n0 * D_MODEL;

#if HAVE_TDM
  unsigned int aoff[2] = {(unsigned int)(size_t)&As[0][0],
                          (unsigned int)(size_t)&As[1][0]};
  unsigned int boff[2] = {(unsigned int)(size_t)&Bs[0][0],
                          (unsigned int)(size_t)&Bs[1][0]};
  if (wid == 0) {
    tdm_load_tile(aoff[0], Arow);
    tdm_load_tile(boff[0], Brow);
  }
#endif
  int p = 0;
  for (int k0 = 0; k0 < D_MODEL; k0 += 32) {
#if HAVE_TDM
    if (wid == 0) __builtin_amdgcn_s_wait_tensorcnt(0);
    __syncthreads();
    if (wid == 0 && (k0 + 32) < D_MODEL) {
      tdm_load_tile(aoff[p ^ 1], Arow + k0 + 32);
      tdm_load_tile(boff[p ^ 1], Brow + k0 + 32);
    }
#else
    const int lrow = t >> 1, lseg = t & 1;
    const v8h* pa_ = (const v8h*)(Arow + (size_t)lrow * D_MODEL + k0 + lseg * 16);
    v8h a0 = pa_[0], a1 = pa_[1];
    const v8h* pb_ = (const v8h*)(Brow + (size_t)lrow * D_MODEL + k0 + lseg * 16);
    v8h b0 = pb_[0], b1 = pb_[1];
    __syncthreads();
    *(v8h*)(&As[0][lrow * LDP + lseg * 16]) = a0;
    *(v8h*)(&As[0][lrow * LDP + lseg * 16 + 8]) = a1;
    *(v8h*)(&Bs[0][lrow * LDP + lseg * 16]) = b0;
    *(v8h*)(&Bs[0][lrow * LDP + lseg * 16 + 8]) = b1;
    __syncthreads();
#endif
    const f16* Asb = &As[p][0];
    const f16* Bsb = &Bs[p][0];
    v16h af[4], bf[2];
#pragma unroll
    for (int mt = 0; mt < 4; ++mt)
      af[mt] = make_a(Asb + (wm * 64 + mt * 16 + ln) * LDP + half * 8);
#pragma unroll
    for (int nt = 0; nt < 2; ++nt)
      bf[nt] = make_b(Bsb + (wn * 32 + nt * 16 + ln) * LDP + half * 8);
#pragma unroll
    for (int mt = 0; mt < 4; ++mt)
#pragma unroll
      for (int nt = 0; nt < 2; ++nt)
        acc[mt][nt] = wmma16(af[mt], bf[nt], acc[mt][nt]);
#if HAVE_TDM
    p ^= 1;
#endif
  }

#pragma unroll
  for (int mt = 0; mt < 4; ++mt) {
#pragma unroll
    for (int nt = 0; nt < 2; ++nt) {
      int gn = n0 + wn * 32 + nt * 16 + ln;
      float bb = bias[gn];
      int which = gn >> 10;        // 0=q 1=k 2=v
      int h = (gn >> 6) & 15;
      int d = gn & 63;
#pragma unroll
      for (int r = 0; r < 8; ++r) {
        int gm = m0 + wm * 64 + mt * 16 + half * 8 + r;
        int b = gm >> 11;
        int tt = gm & (T_SEQ - 1);
        float val = acc[mt][nt][r] + bb;
        size_t bh = (size_t)(b * NHEAD + h);
        if (which == 0)
          Q[(bh * T_SEQ + tt) * HDIM + d] = (f16)(val * 0.125f);  // fold hd^-0.5
        else if (which == 1)
          Kt[(bh * T_SEQ + tt) * HDIM + d] = (f16)val;
        else
          Vt[(bh * HDIM + d) * T_SEQ + tt] = (f16)val;
      }
    }
  }
}

// ---------- flash attention: one wave = 16 query rows, online softmax ----------

__global__ __launch_bounds__(256) void attn_kernel(
    const f16* __restrict__ Q, const f16* __restrict__ Kt,
    const f16* __restrict__ Vt, const int* __restrict__ mask,
    f16* __restrict__ O) {
  __shared__ __attribute__((aligned(16))) f16 plds[8][16 * LDP];
  const int bh = blockIdx.x >> 4;   // 0..31
  const int qblk = blockIdx.x & 15;
  const int lane = threadIdx.x & 31, wid = threadIdx.x >> 5;
  const int half = lane >> 4, ln = lane & 15;
  const int b = bh >> 4, h = bh & 15;
  const int q0 = qblk * 128 + wid * 16;

  // Q fragments for contraction chunks d=[0,32) and [32,64)
  const f16* qbase = Q + ((size_t)bh * T_SEQ + q0 + ln) * HDIM;
  v16h qa0 = make_a(qbase + 0 + half * 8);
  v16h qa1 = make_a(qbase + 32 + half * 8);

  float mrow[8], lrow[8];
#pragma unroll
  for (int r = 0; r < 8; ++r) { mrow[r] = -1e30f; lrow[r] = 0.f; }
  v8f oacc[4];
#pragma unroll
  for (int t4 = 0; t4 < 4; ++t4) oacc[t4] = vzero8();

  f16* pbuf = &plds[wid][0];

  for (int kc = 0; kc < T_SEQ; kc += 32) {
    // S = Q K^T for 32 keys (two 16-key column tiles)
    v8f s0 = vzero8(), s1 = vzero8();
    const f16* kb0 = Kt + ((size_t)bh * T_SEQ + kc + ln) * HDIM + half * 8;
    s0 = wmma16(qa0, make_b(kb0), s0);
    s0 = wmma16(qa1, make_b(kb0 + 32), s0);
    const f16* kb1 = Kt + ((size_t)bh * T_SEQ + kc + 16 + ln) * HDIM + half * 8;
    s1 = wmma16(qa0, make_b(kb1), s1);
    s1 = wmma16(qa1, make_b(kb1 + 32), s1);

    // key mask (reference: score = -inf where mask==0)
    int mk0 = mask[b * T_SEQ + kc + ln];
    int mk1 = mask[b * T_SEQ + kc + 16 + ln];
#pragma unroll
    for (int r = 0; r < 8; ++r) {
      s0[r] = mk0 ? s0[r] : -1e30f;
      s1[r] = mk1 ? s1[r] : -1e30f;
    }

    // online softmax: row stats via cross-lane reduction over the 16-lane half
    float mnew[8], fac[8];
#pragma unroll
    for (int r = 0; r < 8; ++r) {
      float tmx = fmaxf(s0[r], s1[r]);
      tmx = fmaxf(tmx, __shfl_xor(tmx, 1));
      tmx = fmaxf(tmx, __shfl_xor(tmx, 2));
      tmx = fmaxf(tmx, __shfl_xor(tmx, 4));
      tmx = fmaxf(tmx, __shfl_xor(tmx, 8));
      mnew[r] = fmaxf(mrow[r], tmx);
      fac[r] = __expf(mrow[r] - mnew[r]);
      mrow[r] = mnew[r];
    }
#pragma unroll
    for (int r = 0; r < 8; ++r) {
      float p0 = __expf(s0[r] - mnew[r]);
      float p1 = __expf(s1[r] - mnew[r]);
      float rs = p0 + p1;
      rs += __shfl_xor(rs, 1);
      rs += __shfl_xor(rs, 2);
      rs += __shfl_xor(rs, 4);
      rs += __shfl_xor(rs, 8);
      lrow[r] = lrow[r] * fac[r] + rs;
      // transpose P (C layout -> A layout) through per-wave LDS scratch
      pbuf[(half * 8 + r) * LDP + ln] = (f16)p0;
      pbuf[(half * 8 + r) * LDP + 16 + ln] = (f16)p1;
    }
#pragma unroll
    for (int t4 = 0; t4 < 4; ++t4)
#pragma unroll
      for (int r = 0; r < 8; ++r) oacc[t4][r] *= fac[r];

    // O += P V : A = P (16x32), B = V^T rows are contiguous in keys
    v16h pa = make_a(pbuf + ln * LDP + half * 8);
#pragma unroll
    for (int t4 = 0; t4 < 4; ++t4) {
      const f16* vb =
          Vt + ((size_t)bh * HDIM + t4 * 16 + ln) * T_SEQ + kc + half * 8;
      oacc[t4] = wmma16(pa, make_b(vb), oacc[t4]);
    }
  }

  // normalize and store attention output in [B,T,D] f16 for the out-proj GEMM
#pragma unroll
  for (int r = 0; r < 8; ++r) lrow[r] = 1.0f / lrow[r];
#pragma unroll
  for (int t4 = 0; t4 < 4; ++t4) {
#pragma unroll
    for (int r = 0; r < 8; ++r) {
      int row = q0 + half * 8 + r;
      O[((size_t)b * T_SEQ + row) * D_MODEL + h * HDIM + t4 * 16 + ln] =
          (f16)(oacc[t4][r] * lrow[r]);
    }
  }
}

// ---------- output projection GEMM: [4096 x 1024] x [1024 x 1024] + b ----------

__global__ __launch_bounds__(256) void gemm_out_kernel(
    const f16* __restrict__ A, const f16* __restrict__ WT,
    const float* __restrict__ bias, float* __restrict__ out) {
  __shared__ __attribute__((aligned(16))) f16 As[2][128 * LDP];
  __shared__ __attribute__((aligned(16))) f16 Bs[2][128 * LDP];
  const int m0 = blockIdx.y * 128;
  const int n0 = blockIdx.x * 128;
  const int t = threadIdx.x;
  const int lane = t & 31, wid = t >> 5;
  const int wm = wid & 1, wn = wid >> 1;
  const int half = lane >> 4, ln = lane & 15;

  v8f acc[4][2];
#pragma unroll
  for (int i = 0; i < 4; ++i)
#pragma unroll
    for (int j = 0; j < 2; ++j) acc[i][j] = vzero8();

  const f16* Arow = A + (size_t)m0 * D_MODEL;
  const f16* Brow = WT + (size_t)n0 * D_MODEL;

#if HAVE_TDM
  unsigned int aoff[2] = {(unsigned int)(size_t)&As[0][0],
                          (unsigned int)(size_t)&As[1][0]};
  unsigned int boff[2] = {(unsigned int)(size_t)&Bs[0][0],
                          (unsigned int)(size_t)&Bs[1][0]};
  if (wid == 0) {
    tdm_load_tile(aoff[0], Arow);
    tdm_load_tile(boff[0], Brow);
  }
#endif
  int p = 0;
  for (int k0 = 0; k0 < D_MODEL; k0 += 32) {
#if HAVE_TDM
    if (wid == 0) __builtin_amdgcn_s_wait_tensorcnt(0);
    __syncthreads();
    if (wid == 0 && (k0 + 32) < D_MODEL) {
      tdm_load_tile(aoff[p ^ 1], Arow + k0 + 32);
      tdm_load_tile(boff[p ^ 1], Brow + k0 + 32);
    }
#else
    const int lrow = t >> 1, lseg = t & 1;
    const v8h* pa_ = (const v8h*)(Arow + (size_t)lrow * D_MODEL + k0 + lseg * 16);
    v8h a0 = pa_[0], a1 = pa_[1];
    const v8h* pb_ = (const v8h*)(Brow + (size_t)lrow * D_MODEL + k0 + lseg * 16);
    v8h b0 = pb_[0], b1 = pb_[1];
    __syncthreads();
    *(v8h*)(&As[0][lrow * LDP + lseg * 16]) = a0;
    *(v8h*)(&As[0][lrow * LDP + lseg * 16 + 8]) = a1;
    *(v8h*)(&Bs[0][lrow * LDP + lseg * 16]) = b0;
    *(v8h*)(&Bs[0][lrow * LDP + lseg * 16 + 8]) = b1;
    __syncthreads();
#endif
    const f16* Asb = &As[p][0];
    const f16* Bsb = &Bs[p][0];
    v16h af[4], bf[2];
#pragma unroll
    for (int mt = 0; mt < 4; ++mt)
      af[mt] = make_a(Asb + (wm * 64 + mt * 16 + ln) * LDP + half * 8);
#pragma unroll
    for (int nt = 0; nt < 2; ++nt)
      bf[nt] = make_b(Bsb + (wn * 32 + nt * 16 + ln) * LDP + half * 8);
#pragma unroll
    for (int mt = 0; mt < 4; ++mt)
#pragma unroll
      for (int nt = 0; nt < 2; ++nt)
        acc[mt][nt] = wmma16(af[mt], bf[nt], acc[mt][nt]);
#if HAVE_TDM
    p ^= 1;
#endif
  }

#pragma unroll
  for (int mt = 0; mt < 4; ++mt) {
#pragma unroll
    for (int nt = 0; nt < 2; ++nt) {
      int gn = n0 + wn * 32 + nt * 16 + ln;
      float bb = bias[gn];
#pragma unroll
      for (int r = 0; r < 8; ++r) {
        int gm = m0 + wm * 64 + mt * 16 + half * 8 + r;
        out[(size_t)gm * D_MODEL + gn] = acc[mt][nt][r] + bb;
      }
    }
  }
}

// ---------- launch ----------

extern "C" void kernel_launch(void* const* d_in, const int* in_sizes, int n_in,
                              void* d_out, int out_size, void* d_ws,
                              size_t ws_size, hipStream_t stream) {
  const float* x = (const float*)d_in[0];
  const int* mask = (const int*)d_in[1];
  const float* W_qkv = (const float*)d_in[2];
  const float* b_qkv = (const float*)d_in[3];
  const float* W_out = (const float*)d_in[4];
  const float* b_out = (const float*)d_in[5];
  float* out = (float*)d_out;

  char* ws = (char*)d_ws;
  size_t off = 0;
  auto take = [&](size_t elems) {
    void* p = ws + off;
    off += elems * sizeof(f16);
    return p;
  };
  const size_t QKV_ELEMS = (size_t)NBATCH * NHEAD * T_SEQ * HDIM;  // 4M
  f16* Xb = (f16*)take((size_t)M_ROWS * D_MODEL);   // x in f16
  f16* WqT = (f16*)take((size_t)N_QKV * D_MODEL);   // W_qkv^T f16
  f16* WoT = (f16*)take((size_t)D_MODEL * D_MODEL); // W_out^T f16
  f16* Qb = (f16*)take(QKV_ELEMS);                  // [B,H,T,64]
  f16* Kb = (f16*)take(QKV_ELEMS);                  // [B,H,T,64]
  f16* Vt = (f16*)take(QKV_ELEMS);                  // [B,H,64,T]
  f16* Ab = (f16*)take((size_t)M_ROWS * D_MODEL);   // attn out [B,T,D]

  cvt_f16_kernel<<<1024, 256, 0, stream>>>(x, Xb, M_ROWS * D_MODEL);
  cvt_trans_f16_kernel<<<1024, 256, 0, stream>>>(W_qkv, WqT, D_MODEL, N_QKV);
  cvt_trans_f16_kernel<<<512, 256, 0, stream>>>(W_out, WoT, D_MODEL, D_MODEL);

  gemm_qkv_kernel<<<dim3(N_QKV / 128, M_ROWS / 128), 256, 0, stream>>>(
      Xb, WqT, b_qkv, Qb, Kb, Vt);

  attn_kernel<<<NBATCH * NHEAD * (T_SEQ / 128), 256, 0, stream>>>(
      Qb, Kb, Vt, mask, Ab);

  gemm_out_kernel<<<dim3(D_MODEL / 128, M_ROWS / 128), 256, 0, stream>>>(
      Ab, WoT, b_out, out);
}
